// CollisionGridModel_11776800325718
// MI455X (gfx1250) — matrix operand order repaired and checked by reading, not compiled, not run on hardware
//
#include <hip/hip_runtime.h>

// ---- problem constants (from reference) ----
#define T_FR   19
#define N_PED  512
#define V_VEH  64
#define S_SOC  24     // 3*8 sectors
#define E_EMB  128
#define R_RNN  256
#define O_OUT  5
#define K_X    384    // 3*E
#define K_H    256    // R
#define K_TOT  640    // K_X + K_H
#define G_COLS 1024   // 4*R
#define KT_TOT (K_TOT / 32)   // 20 k-tiles of 32
#define NT_TOT (G_COLS / 16)  // 64 col tiles

typedef __attribute__((ext_vector_type(16))) __bf16 v16bf;
typedef __attribute__((ext_vector_type(8)))  float  v8f;

struct alignas(16) U128 { unsigned int w[4]; };     // trivial 16B chunk
union Frag { U128 q[2]; v16bf v; };                 // 32B = one bf16 WMMA operand/lane

static __device__ __forceinline__ unsigned short f32_to_bf16(float f) {
  unsigned int u = __float_as_uint(f);
  if ((u & 0x7F800000u) == 0x7F800000u) return (unsigned short)(u >> 16); // inf/nan
  unsigned int r = u + 0x7FFFu + ((u >> 16) & 1u);                        // RNE
  return (unsigned short)(r >> 16);
}
static __device__ __forceinline__ float sigmoidf(float x) {
  return 1.0f / (1.0f + __expf(-x));
}

// ---------------------------------------------------------------------------
// Max-pool over the "other agent" axis: dst[t*N+i, s] = max_j src[t*N+i, j, s]
// One block per (t,i); 192 threads = 24 sectors x 8 j-groups. Each loop
// iteration reads 768B contiguous per block -> fully coalesced stream of the
// 538MB TTC grids (the HBM-roofline bulk: ~23us at 23.3TB/s).
// ---------------------------------------------------------------------------
__global__ __launch_bounds__(192)
void maxpool_kernel(const float* __restrict__ src, float* __restrict__ dst, int J) {
  __shared__ float red[192];
  const int s  = threadIdx.x % S_SOC;
  const int jj = threadIdx.x / S_SOC;
  const size_t rowBase = (size_t)blockIdx.x * (size_t)J * S_SOC;
  float m = -3.4e38f;
  for (int j = jj; j < J; j += 8)
    m = fmaxf(m, src[rowBase + (size_t)j * S_SOC + s]);
  red[threadIdx.x] = m;
  __syncthreads();
  if (jj == 0) {
    float r = red[s];
#pragma unroll
    for (int k = 1; k < 8; ++k) r = fmaxf(r, red[k * S_SOC + s]);
    dst[(size_t)blockIdx.x * S_SOC + s] = r;
  }
}

// ---------------------------------------------------------------------------
// Embeddings for ALL frames in parallel (no h dependence):
// x[t*N+n, 0:384] = [relu(nodes@W_in+b) | relu(social@W_t+b) | relu(socv@W_tv+b)]
// K is tiny (6/24/24): plain VALU dots; output stored bf16 for the WMMA GEMM.
// ---------------------------------------------------------------------------
__global__ __launch_bounds__(256)
void embed_kernel(const float* __restrict__ input_data,
                  const float* __restrict__ social,
                  const float* __restrict__ social_v,
                  const float* __restrict__ W_in, const float* __restrict__ b_in,
                  const float* __restrict__ W_t,  const float* __restrict__ b_t,
                  const float* __restrict__ W_tv, const float* __restrict__ b_tv,
                  unsigned short* __restrict__ Xb) {
  int idx = blockIdx.x * blockDim.x + threadIdx.x;
  if (idx >= T_FR * N_PED * E_EMB) return;
  const int j   = idx % E_EMB;
  const int row = idx / E_EMB;                     // t*N + n
  const float* f = input_data + (size_t)row * 9;   // frame features [9]
  const float nd[6] = { f[0], f[1], f[5], f[6], f[7], f[8] };
  float a = b_in[j];
#pragma unroll
  for (int k = 0; k < 6; ++k) a += nd[k] * W_in[k * E_EMB + j];
  float b = b_t[j];
  const float* sc = social + (size_t)row * S_SOC;
#pragma unroll
  for (int k = 0; k < S_SOC; ++k) b += sc[k] * W_t[k * E_EMB + j];
  float c = b_tv[j];
  const float* scv = social_v + (size_t)row * S_SOC;
#pragma unroll
  for (int k = 0; k < S_SOC; ++k) c += scv[k] * W_tv[k * E_EMB + j];
  unsigned short* xr = Xb + (size_t)row * K_X;
  xr[j]             = f32_to_bf16(fmaxf(a, 0.0f));
  xr[E_EMB + j]     = f32_to_bf16(fmaxf(b, 0.0f));
  xr[2 * E_EMB + j] = f32_to_bf16(fmaxf(c, 0.0f));
}

// ---------------------------------------------------------------------------
// One-time prep:
//  (1) Pack fused W = [W_ih ; W_hh] (640x1024 f32) into bf16 *WMMA B-fragment
//      order*: Wp[kt][nt][lane][e], e=0..15, so each lane's whole B operand is
//      32B contiguous (two global_load_b128 instead of 16 strided u16 loads).
//      B layout (CDNA5 ISA 7.12.2): lane n = L&15, half = L>>4,
//      element e -> k = kt*32 + half*8 + (e&7) + (e>>3)*16.
//  (2) Seed f32 h/c state and bf16 h from the inputs.
// ---------------------------------------------------------------------------
__global__ __launch_bounds__(256)
void prep_kernel(const float* __restrict__ W_ih, const float* __restrict__ W_hh,
                 const float* __restrict__ h0, const float* __restrict__ c0,
                 unsigned short* __restrict__ Wp, float* __restrict__ h,
                 float* __restrict__ c, unsigned short* __restrict__ hb) {
  int idx = blockIdx.x * blockDim.x + threadIdx.x;
  if (idx < KT_TOT * NT_TOT * 32 * 16) {
    const int e    = idx & 15;
    const int lane = (idx >> 4) & 31;
    const int nt   = (idx >> 9) & 63;
    const int kt   = idx >> 15;
    const int hf   = lane >> 4;
    const int n = (nt << 4) + (lane & 15);
    const int k = (kt << 5) + (hf << 3) + (e & 7) + ((e >> 3) << 4);
    const float w = (k < K_X) ? W_ih[(size_t)k * G_COLS + n]
                              : W_hh[(size_t)(k - K_X) * G_COLS + n];
    Wp[idx] = f32_to_bf16(w);
  }
  if (idx < N_PED * R_RNN) {
    const float hv = h0[idx];
    h[idx]  = hv;
    c[idx]  = c0[idx];
    hb[idx] = f32_to_bf16(hv);
  }
}

// ---------------------------------------------------------------------------
// Per-step fused gate GEMM: gates[512,1024] = [x_t | h] @ W[640,1024] + biases
// One wave per 16x32 output strip (two 16x16 WMMA tiles sharing the A
// fragment) -> 1024 waves = 128 blocks x 8 waves. K split: 12 k-tiles over x
// (ld=384), 8 over h (ld=256). bf16 V_WMMA_F32_16X16X32_BF16, f32 accumulate.
// A fragment is two contiguous b128 loads/lane; B fragment is two contiguous
// b128 loads/lane from the pre-packed Wp.
// ---------------------------------------------------------------------------
__global__ __launch_bounds__(256)
void gates_gemm_kernel(const unsigned short* __restrict__ Xb,  // [N,384] frame t
                       const unsigned short* __restrict__ Hb,  // [N,256]
                       const unsigned short* __restrict__ Wp,  // packed [20][64][32][16]
                       const float* __restrict__ b_ih, const float* __restrict__ b_hh,
                       float* __restrict__ gates) {
  const int wave = (blockIdx.x * 256 + threadIdx.x) >> 5;  // 0..1023
  const int lane = threadIdx.x & 31;
  const int mt  = wave >> 5;          // 32 row tiles
  const int np  = wave & 31;          // 32 column-tile pairs
  const int nt0 = np << 1;
  const int m0  = mt << 4;
  const int hf  = lane >> 4;
  const int arow  = m0 + (lane & 15);
  const int ncol0 = (nt0 << 4) + (lane & 15);
  const int ncol1 = ncol0 + 16;

  v8f acc0 = {}, acc1 = {};

  // ---- K over x part (K = 384, k-tiles 0..11) ----
  const unsigned short* Arow = Xb + (size_t)arow * K_X;
#pragma unroll 1
  for (int kt = 0; kt < K_X / 32; ++kt) {
    const int kb = kt << 5;
    __builtin_prefetch(Arow + kb + 32, 0, 0);   // speculative: next A k-tile
    Frag a, b0, b1;
    const unsigned short* ap = Arow + kb + (hf << 3);
    a.q[0] = *(const U128*)(ap);
    a.q[1] = *(const U128*)(ap + 16);
    const unsigned short* wp = Wp + ((((size_t)kt << 6) + nt0) << 9) + (lane << 4);
    b0.q[0] = *(const U128*)(wp);
    b0.q[1] = *(const U128*)(wp + 8);
    b1.q[0] = *(const U128*)(wp + 512);
    b1.q[1] = *(const U128*)(wp + 520);
    acc0 = __builtin_amdgcn_wmma_f32_16x16x32_bf16(false, a.v, false, b0.v,
                                                   (short)0, acc0, false, false);
    acc1 = __builtin_amdgcn_wmma_f32_16x16x32_bf16(false, a.v, false, b1.v,
                                                   (short)0, acc1, false, false);
  }

  // ---- K over h part (K = 256, k-tiles 12..19) ----
  const unsigned short* Hrow = Hb + (size_t)arow * K_H;
#pragma unroll 1
  for (int kt = 0; kt < K_H / 32; ++kt) {
    const int kb = kt << 5;
    Frag a, b0, b1;
    const unsigned short* ap = Hrow + kb + (hf << 3);
    a.q[0] = *(const U128*)(ap);
    a.q[1] = *(const U128*)(ap + 16);
    const unsigned short* wp =
        Wp + ((((size_t)(kt + K_X / 32) << 6) + nt0) << 9) + (lane << 4);
    b0.q[0] = *(const U128*)(wp);
    b0.q[1] = *(const U128*)(wp + 8);
    b1.q[0] = *(const U128*)(wp + 512);
    b1.q[1] = *(const U128*)(wp + 520);
    acc0 = __builtin_amdgcn_wmma_f32_16x16x32_bf16(false, a.v, false, b0.v,
                                                   (short)0, acc0, false, false);
    acc1 = __builtin_amdgcn_wmma_f32_16x16x32_bf16(false, a.v, false, b1.v,
                                                   (short)0, acc1, false, false);
  }

  // ---- epilogue: D element (m = r + 8*half, n = lane&15) + both biases ----
  const float bias0 = b_ih[ncol0] + b_hh[ncol0];
  const float bias1 = b_ih[ncol1] + b_hh[ncol1];
#pragma unroll
  for (int r = 0; r < 8; ++r) {
    const int m = m0 + r + (hf << 3);
    gates[(size_t)m * G_COLS + ncol0] = acc0[r] + bias0;
    gates[(size_t)m * G_COLS + ncol1] = acc1[r] + bias1;
  }
}

// ---------------------------------------------------------------------------
// LSTM pointwise (torch gate order i,f,g,o in column blocks of 256).
// Writes f32 h/c state + bf16 h for the next step's GEMM; on the last step
// also writes the h,c sections of d_out.
// ---------------------------------------------------------------------------
__global__ __launch_bounds__(256)
void lstm_pointwise_kernel(const float* __restrict__ gates, float* __restrict__ h,
                           float* __restrict__ c, unsigned short* __restrict__ hb,
                           float* __restrict__ outH, float* __restrict__ outC,
                           int last) {
  int idx = blockIdx.x * blockDim.x + threadIdx.x;
  if (idx >= N_PED * R_RNN) return;
  const int n = idx / R_RNN, j = idx % R_RNN;
  const float* g = gates + (size_t)n * G_COLS;
  const float iv = sigmoidf(g[j]);
  const float fv = sigmoidf(g[R_RNN + j]);
  const float gv = tanhf(g[2 * R_RNN + j]);
  const float ov = sigmoidf(g[3 * R_RNN + j]);
  const float c2 = fv * c[idx] + iv * gv;
  const float h2 = ov * tanhf(c2);
  c[idx]  = c2;
  h[idx]  = h2;
  hb[idx] = f32_to_bf16(h2);
  if (last) { outH[idx] = h2; outC[idx] = c2; }
}

// ---------------------------------------------------------------------------
// out[t,n,:] = h2[n,:] @ W_out[256,5] + b_out  (tiny; one thread per output)
// ---------------------------------------------------------------------------
__global__ __launch_bounds__(256)
void outproj_kernel(const float* __restrict__ h, const float* __restrict__ W_out,
                    const float* __restrict__ b_out, float* __restrict__ out) {
  int idx = blockIdx.x * blockDim.x + threadIdx.x;
  if (idx >= N_PED * O_OUT) return;
  const int n = idx / O_OUT, o = idx % O_OUT;
  const float* hr = h + (size_t)n * R_RNN;
  float acc = b_out[o];
  for (int k = 0; k < R_RNN; ++k) acc += hr[k] * W_out[k * O_OUT + o];
  out[idx] = acc;
}

// ---------------------------------------------------------------------------
extern "C" void kernel_launch(void* const* d_in, const int* in_sizes, int n_in,
                              void* d_out, int out_size, void* d_ws, size_t ws_size,
                              hipStream_t stream) {
  (void)in_sizes; (void)n_in; (void)out_size; (void)ws_size;

  const float* input_data = (const float*)d_in[0];
  const float* h0         = (const float*)d_in[2];
  const float* c0         = (const float*)d_in[3];
  const float* gTTC       = (const float*)d_in[8];
  const float* gTTCv      = (const float*)d_in[9];
  const float* W_in  = (const float*)d_in[10];
  const float* b_in  = (const float*)d_in[11];
  const float* W_t   = (const float*)d_in[12];
  const float* b_t   = (const float*)d_in[13];
  const float* W_tv  = (const float*)d_in[14];
  const float* b_tv  = (const float*)d_in[15];
  const float* W_ih  = (const float*)d_in[16];
  const float* b_ih  = (const float*)d_in[17];
  const float* W_hh  = (const float*)d_in[18];
  const float* b_hh  = (const float*)d_in[19];
  const float* W_out = (const float*)d_in[20];
  const float* b_out = (const float*)d_in[21];

  // ---- workspace carve-out (~14 MB), 256B-aligned slices ----
  char* ws = (char*)d_ws;
  size_t off = 0;
  auto take = [&](size_t bytes) -> char* {
    char* p = ws + off;
    off = (off + bytes + 255) & ~(size_t)255;
    return p;
  };
  float*          social   = (float*)take((size_t)T_FR * N_PED * S_SOC * 4);
  float*          social_v = (float*)take((size_t)T_FR * N_PED * S_SOC * 4);
  unsigned short* Xb       = (unsigned short*)take((size_t)T_FR * N_PED * K_X * 2);
  unsigned short* Wp       = (unsigned short*)take((size_t)K_TOT * G_COLS * 2);
  float*          gates    = (float*)take((size_t)N_PED * G_COLS * 4);
  float*          h        = (float*)take((size_t)N_PED * R_RNN * 4);
  float*          c        = (float*)take((size_t)N_PED * R_RNN * 4);
  unsigned short* hb       = (unsigned short*)take((size_t)N_PED * R_RNN * 2);

  // d_out = [outs (T*N*5) | h (N*256) | c (N*256)] f32
  float* out_seq = (float*)d_out;
  float* outH = out_seq + (size_t)T_FR * N_PED * O_OUT;
  float* outC = outH + (size_t)N_PED * R_RNN;

  // Phase 1: parallel over all frames (HBM-bound bulk + embeddings + prep)
  maxpool_kernel<<<T_FR * N_PED, 192, 0, stream>>>(gTTC,  social,   N_PED);
  maxpool_kernel<<<T_FR * N_PED, 192, 0, stream>>>(gTTCv, social_v, V_VEH);
  {
    const int total = T_FR * N_PED * E_EMB;
    embed_kernel<<<(total + 255) / 256, 256, 0, stream>>>(
        input_data, social, social_v, W_in, b_in, W_t, b_t, W_tv, b_tv, Xb);
  }
  prep_kernel<<<(K_TOT * G_COLS + 255) / 256, 256, 0, stream>>>(
      W_ih, W_hh, h0, c0, Wp, h, c, hb);

  // Phase 2: sequential LSTM chain (19 steps, each a 512x1024 K=640 bf16 GEMM)
  for (int t = 0; t < T_FR; ++t) {
    gates_gemm_kernel<<<128, 256, 0, stream>>>(
        Xb + (size_t)t * N_PED * K_X, hb, Wp, b_ih, b_hh, gates);
    lstm_pointwise_kernel<<<(N_PED * R_RNN + 255) / 256, 256, 0, stream>>>(
        gates, h, c, hb, outH, outC, (t == T_FR - 1) ? 1 : 0);
    outproj_kernel<<<(N_PED * O_OUT + 255) / 256, 256, 0, stream>>>(
        h, W_out, b_out, out_seq + (size_t)t * N_PED * O_OUT);
  }
}
